// M3LNet_34943853920349
// MI455X (gfx1250) — matrix operand order repaired and checked by reading, not compiled, not run on hardware
//
#include <hip/hip_runtime.h>
#include <hip/hip_bf16.h>

typedef __attribute__((ext_vector_type(16))) _Float16 v16h;
typedef __attribute__((ext_vector_type(8)))  _Float16 v8h;
typedef __attribute__((ext_vector_type(8)))  float    v8f;

// ---------------- problem constants ----------------
#define NB    32
#define NPG   290
#define NNODE 9280
#define NEDGE 74240
#define LP    900
#define INF_  75
#define INFP  96      /* 75 padded to mult of 32 */
#define DD    128
#define HKC   768
#define HDC   256
#define L1C   898
#define L2C   893
#define L3C   885
#define L3P   896     /* 885 padded (K role) / ceil64 (N role) */

static inline unsigned cdiv(long a, long b) { return (unsigned)((a + b - 1) / b); }

// ---------------- generic WMMA GEMM (NT, fully padded: no masking in K-loop) ----------------
// C[b](MxN) = epilogue( A[b](M x Kp, f16, row stride lda) @ Bt[b](ceil64(N) x Kp, f16)^T )
// Kp multiple of 32; Bt zero-filled in padding; A M-overrun rows are readable slack.
// Optional per-row remap of A for implicit im2col: arow = m + (m/rowMod)*rowSkip.
// Epilogue: + bias[n] + sbias[0], optional ReLU, optional += Cacc, writes f32 and/or f16.
struct GemmP {
  const _Float16* A; const _Float16* Bt; float* C32; _Float16* C16;
  const float* bias; const float* sbias; const float* Cacc;
  long sA, sB, sC;
  int lda, ldb, ldc, rowMod, rowSkip, relu, M, N, K;
};

__global__ __launch_bounds__(128) void k_gemm(GemmP p) {
  const int lane  = threadIdx.x & 31;
  const int wave  = threadIdx.x >> 5;
  const int mBase = (blockIdx.x * 4 + wave) * 16;
  const int nBase = blockIdx.y * 64;
  if (mBase >= p.M) return;                 // wave-uniform: EXEC all-1s at WMMA
  const long bz = blockIdx.z;
  const int mRow  = mBase + (lane & 15);
  const int khalf = lane >> 4;
  long arow = mRow;
  if (p.rowMod) arow += (long)(mRow / p.rowMod) * p.rowSkip;
  const _Float16* Arow  = p.A  + bz * p.sA + arow * (long)p.lda + khalf * 8;
  const _Float16* Brow0 = p.Bt + bz * p.sB + (long)(nBase + (lane & 15)) * p.ldb + khalf * 8;
  const long bstep = (long)p.ldb * 16;

  v8f acc[4];
  #pragma unroll
  for (int t = 0; t < 4; ++t)
    #pragma unroll
    for (int r = 0; r < 8; ++r) acc[t][r] = 0.0f;

  for (int kb = 0; kb < p.K; kb += 32) {
    // A fragment 16x32: lanes 0-15 hold K {kb..kb+7, kb+16..kb+23}, lanes 16-31 the +8 halves
    v8h alo = *(const v8h*)(Arow + kb);
    v8h ahi = *(const v8h*)(Arow + kb + 16);
    v16h a = __builtin_shufflevector(alo, ahi, 0, 1, 2, 3, 4, 5, 6, 7,
                                     8, 9, 10, 11, 12, 13, 14, 15);
    __builtin_prefetch(Arow + kb + 256, 0, 1);     // global_prefetch_b8
    #pragma unroll
    for (int t = 0; t < 4; ++t) {
      // B fragment 32x16: lane holds column n, K {kb..kb+7, kb+16..kb+23} (+8 for lanes 16-31) of Bt row n
      v8h blo = *(const v8h*)(Brow0 + (long)t * bstep + kb);
      v8h bhi = *(const v8h*)(Brow0 + (long)t * bstep + kb + 16);
      v16h b = __builtin_shufflevector(blo, bhi, 0, 1, 2, 3, 4, 5, 6, 7,
                                       8, 9, 10, 11, 12, 13, 14, 15);
      acc[t] = __builtin_amdgcn_wmma_f32_16x16x32_f16(false, a, false, b,
                                                      (short)0, acc[t], false, false);
    }
  }

  const float sb = p.sbias ? p.sbias[0] : 0.0f;
  #pragma unroll
  for (int t = 0; t < 4; ++t) {
    const int n = nBase + t * 16 + (lane & 15);
    if (n >= p.N) continue;
    const float badd = sb + (p.bias ? p.bias[n] : 0.0f);
    #pragma unroll
    for (int r = 0; r < 8; ++r) {
      const int m = mBase + khalf * 8 + r;
      if (m >= p.M) continue;
      float v = acc[t][r] + badd;
      if (p.relu) v = fmaxf(v, 0.0f);
      const long ci = bz * p.sC + (long)m * p.ldc + n;
      if (p.Cacc) v += p.Cacc[ci];
      if (p.C32) p.C32[ci] = v;
      if (p.C16) p.C16[ci] = (_Float16)v;
    }
  }
}

static inline void gemm(hipStream_t st,
                        const _Float16* A, long sA, int lda, int rowMod, int rowSkip,
                        const _Float16* Bt, long sB, int ldb,
                        float* C32, _Float16* C16, long sC, int ldc,
                        const float* bias, const float* sbias, const float* Cacc,
                        int relu, int M, int N, int K, int batch) {
  GemmP p{A, Bt, C32, C16, bias, sbias, Cacc, sA, sB, sC,
          lda, ldb, ldc, rowMod, rowSkip, relu, M, N, K};
  dim3 g(cdiv(M, 64), cdiv(N, 64), (unsigned)batch);
  k_gemm<<<g, dim3(128), 0, st>>>(p);
}

// ---------------- small helper kernels ----------------
__global__ void k_f32_to_f16(const float* __restrict__ s, _Float16* __restrict__ d, long n) {
  long i = (long)blockIdx.x * blockDim.x + threadIdx.x;
  if (i < n) d[i] = (_Float16)s[i];
}
__global__ void k_zero(float* p, long n) {
  long i = (long)blockIdx.x * blockDim.x + threadIdx.x;
  if (i < n) p[i] = 0.0f;
}
__global__ void k_zero16(_Float16* p, long n) {
  long i = (long)blockIdx.x * blockDim.x + threadIdx.x;
  if (i < n) p[i] = (_Float16)0.0f;
}
// f32 [rows,K] -> f16 [rows,Kp], zero-fill K..Kp-1
__global__ void k_cvt_pad(const float* __restrict__ s, _Float16* __restrict__ d,
                          int K, int Kp, long n) {
  long i = (long)blockIdx.x * blockDim.x + threadIdx.x;
  if (i >= n) return;
  int k = (int)(i % Kp); long r = i / Kp;
  d[i] = (k < K) ? (_Float16)s[r * (long)K + k] : (_Float16)0.0f;
}
// f32 src [K,N] -> f16 dst [Np,Kp] transposed, zero-padded
__global__ void k_transpose_pad(const float* __restrict__ s, _Float16* __restrict__ d,
                                int K, int N, int Kp, int Np) {
  long n_ = (long)Np * Kp;
  long i = (long)blockIdx.x * blockDim.x + threadIdx.x;
  if (i >= n_) return;
  int k = (int)(i % Kp); long nn = i / Kp;
  d[i] = (nn < N && k < K) ? (_Float16)s[(long)k * N + nn] : (_Float16)0.0f;
}
__global__ void k_scatter(const float* __restrict__ h, const int* __restrict__ es,
                          const int* __restrict__ ed, float* __restrict__ agg, long n) {
  long i = (long)blockIdx.x * blockDim.x + threadIdx.x;
  if (i >= n) return;
  long e = i >> 7; int d = (int)(i & 127);
  atomicAdd(&agg[(long)ed[e] * DD + d], h[(long)es[e] * DD + d]);
}
// conv weight (Dout, Din, k) -> Bt [dout, t*D+din] f16 (already NT layout)
__global__ void k_convw(const float* __restrict__ src, _Float16* __restrict__ dst, int kk) {
  long Kp = (long)kk * DD;
  long n = Kp * DD;
  long i = (long)blockIdx.x * blockDim.x + threadIdx.x;
  if (i >= n) return;
  int k = (int)(i % Kp); long dout = i / Kp;
  int t = k >> 7, din = k & 127;
  dst[i] = (_Float16)src[(dout * DD + din) * kk + t];
}
__global__ void k_embed(const int* __restrict__ vp, const float* __restrict__ emb,
                        _Float16* __restrict__ x) {
  long n = (long)NB * LP * DD;
  long i = (long)blockIdx.x * blockDim.x + threadIdx.x;
  if (i >= n) return;
  int d = (int)(i & 127); long t = i >> 7;
  int tok = vp[t];
  x[i] = (_Float16)emb[(long)tok * DD + d];
}
// batched transpose: qT[b][k][n] = (n<885) ? q[b][n][k] : 0 ;  q is [NB,896,768], qT [NB,768,896]
__global__ void k_qT(const _Float16* __restrict__ q, _Float16* __restrict__ qT) {
  long n_ = (long)NB * HKC * L3P;
  long i = (long)blockIdx.x * blockDim.x + threadIdx.x;
  if (i >= n_) return;
  int n = (int)(i % L3P); long r = i / L3P;
  int k = (int)(r % HKC); long b = r / HKC;
  qT[i] = (n < L3C) ? q[((long)b * L3P + n) * HKC + k] : (_Float16)0.0f;
}
__global__ void k_bnchan_stats(const float* __restrict__ x, float* __restrict__ stats, int rows) {
  __shared__ float ss[256], sq[256];
  int c = blockIdx.x, t = threadIdx.x;
  float s = 0.f, q = 0.f;
  for (int r = t; r < rows; r += 256) { float v = x[(long)r * DD + c]; s += v; q += v * v; }
  ss[t] = s; sq[t] = q; __syncthreads();
  for (int o = 128; o > 0; o >>= 1) {
    if (t < o) { ss[t] += ss[t + o]; sq[t] += sq[t + o]; }
    __syncthreads();
  }
  if (t == 0) { float m = ss[0] / rows; stats[c] = m; stats[DD + c] = sq[0] / rows - m * m; }
}
__global__ void k_bnchan_apply(const float* __restrict__ x, const float* __restrict__ stats,
                               const float* __restrict__ g, const float* __restrict__ b,
                               _Float16* __restrict__ o, long n) {
  long i = (long)blockIdx.x * blockDim.x + threadIdx.x;
  if (i >= n) return;
  int c = (int)(i & 127);
  float m = stats[c], v = stats[DD + c];
  o[i] = (_Float16)((x[i] - m) * rsqrtf(v + 1e-5f) * g[c] + b[c]);
}
__global__ void k_hsum(const float* __restrict__ h_mat, const float* __restrict__ h_bias,
                       float* __restrict__ hs, float* __restrict__ hb) {
  int i = blockIdx.x * blockDim.x + threadIdx.x;
  if (i < HKC) hs[i] = h_mat[i] + h_mat[HKC + i];
  if (i == 0) hb[0] = h_bias[0] + h_bias[1];
}
__global__ void k_vh(const float* __restrict__ v, const float* __restrict__ hs,
                     _Float16* __restrict__ o, long n) {
  long i = (long)blockIdx.x * blockDim.x + threadIdx.x;
  if (i >= n) return;
  int k = (int)(i % HKC);
  o[i] = (_Float16)(v[i] * hs[k]);
}
__global__ void k_fred(const float* __restrict__ v, const float* __restrict__ T,
                       float* __restrict__ f) {
  int i = blockIdx.x * blockDim.x + threadIdx.x;
  if (i >= NB * HKC) return;
  int b = i / HKC, k = i % HKC;
  long base = ((long)b * NPG) * HKC + k;
  float s = 0.f;
  for (int vv = 0; vv < NPG; ++vv) { s += v[base] * T[base]; base += HKC; }
  f[i] = s;
}
__global__ void k_group3(const float* __restrict__ f, float* __restrict__ f2) {
  int i = blockIdx.x * blockDim.x + threadIdx.x;
  if (i >= NB * HDC) return;
  int b = i / HDC, c = i % HDC;
  long o = (long)b * HKC + c * 3;
  f2[i] = f[o] + f[o + 1] + f[o + 2];
}
__global__ void k_bnfeat(const float* __restrict__ x, const float* __restrict__ g,
                         const float* __restrict__ b, float* o32, _Float16* o16, int C) {
  int c = blockIdx.x * blockDim.x + threadIdx.x;
  if (c >= C) return;
  float s = 0.f, q = 0.f;
  for (int r = 0; r < NB; ++r) { float v = x[(long)r * C + c]; s += v; q += v * v; }
  float m = s / NB, var = q / NB - m * m;
  float sc = rsqrtf(var + 1e-5f) * g[c], bb = b[c];
  for (int r = 0; r < NB; ++r) {
    float v = (x[(long)r * C + c] - m) * sc + bb;
    if (o32) o32[(long)r * C + c] = v;
    if (o16) o16[(long)r * C + c] = (_Float16)v;
  }
}
__global__ void k_dot_out(const float* __restrict__ x, const float* __restrict__ w,
                          const float* __restrict__ b4, float* __restrict__ out) {
  int i = threadIdx.x;
  if (i >= NB) return;
  float s = b4[0];
  for (int j = 0; j < DD; ++j) s += x[(long)i * DD + j] * w[j];
  out[i] = s;
}
__global__ void k_final(const float* __restrict__ sc, float* __restrict__ out) {
  __shared__ float n1[NB], n2[NB], red[NB];
  int i = threadIdx.x;
  float a = sc[NB + i];      n1[i] = a / fmaxf(fabsf(a), 1e-12f);
  float c = sc[2 * NB + i];  n2[i] = c / fmaxf(fabsf(c), 1e-12f);
  __syncthreads();
  float mx = -1e30f;
  for (int j = 0; j < NB; ++j) mx = fmaxf(mx, n1[i] * n2[j]);
  float se = 0.f;
  for (int j = 0; j < NB; ++j) se += expf(n1[i] * n2[j] - mx);
  red[i] = n1[i] * n2[i] - mx - logf(se);
  __syncthreads();
  if (i == 0) { float s = 0.f; for (int j = 0; j < NB; ++j) s += red[j]; red[0] = -s / NB; }
  __syncthreads();
  float loss = red[0];
  out[i * 4 + 0] = sc[i];
  out[i * 4 + 1] = sc[NB + i];
  out[i * 4 + 2] = sc[2 * NB + i];
  out[i * 4 + 3] = loss;
}

// ---------------- orchestration ----------------
extern "C" void kernel_launch(void* const* d_in, const int* in_sizes, int n_in,
                              void* d_out, int out_size, void* d_ws, size_t ws_size,
                              hipStream_t stream) {
  (void)in_sizes; (void)n_in; (void)out_size; (void)ws_size;
  const float* node_h   = (const float*)d_in[0];
  const int*   esrc     = (const int*)d_in[1];
  const int*   edst     = (const int*)d_in[2];
  const int*   v_p      = (const int*)d_in[3];
  const float* W_init   = (const float*)d_in[4];
  const float* gcn_W    = (const float*)d_in[5];
  const float* gcn_b    = (const float*)d_in[6];
  const float* gcn_Wres = (const float*)d_in[7];
  const float* gcn_bres = (const float*)d_in[8];
  const float* emb      = (const float*)d_in[9];
  const float* conv1_w  = (const float*)d_in[10];
  const float* conv2_w  = (const float*)d_in[11];
  const float* conv3_w  = (const float*)d_in[12];
  const float* conv_b   = (const float*)d_in[13];
  const float* cbn_g    = (const float*)d_in[14];
  const float* cbn_b    = (const float*)d_in[15];
  const float* Wv       = (const float*)d_in[16];
  const float* bv       = (const float*)d_in[17];
  const float* Wq       = (const float*)d_in[18];
  const float* bq       = (const float*)d_in[19];
  const float* h_mat    = (const float*)d_in[20];
  const float* h_bias   = (const float*)d_in[21];
  const float* ban_g    = (const float*)d_in[22];
  const float* ban_b    = (const float*)d_in[23];
  const float* m_w1     = (const float*)d_in[24];
  const float* m_b1     = (const float*)d_in[25];
  const float* m_w2     = (const float*)d_in[26];
  const float* m_b2     = (const float*)d_in[27];
  const float* m_w3     = (const float*)d_in[28];
  const float* m_b3     = (const float*)d_in[29];
  const float* m_w4     = (const float*)d_in[30];
  const float* m_b4     = (const float*)d_in[31];
  const float* mg1 = (const float*)d_in[32]; const float* mb1 = (const float*)d_in[33];
  const float* mg2 = (const float*)d_in[34]; const float* mb2 = (const float*)d_in[35];
  const float* mg3 = (const float*)d_in[36]; const float* mb3 = (const float*)d_in[37];
  float* out = (float*)d_out;

  char* wp = (char*)d_ws; size_t off = 0;
  auto alloc = [&](size_t bytes) -> void* {
    void* p = wp + off; off += (bytes + 255) & ~(size_t)255; return p;
  };

  // A-role inputs (padded K / M-overrun slack where needed)
  _Float16* nodeh16 = (_Float16*)alloc((long)NNODE * INFP * 2);
  // Bt weights (transposed, [Np, Kp])
  _Float16* WinitT  = (_Float16*)alloc(128L * INFP * 2);
  _Float16* gcnWT   = (_Float16*)alloc(9L * DD * DD * 2);
  _Float16* gcnWrT  = (_Float16*)alloc(9L * DD * DD * 2);
  _Float16* WvT     = (_Float16*)alloc((long)HKC * DD * 2);
  _Float16* WqT     = (_Float16*)alloc((long)HKC * DD * 2);
  _Float16* cwT     = (_Float16*)alloc((long)DD * (3 + 6 + 9) * DD * 2);
  _Float16* mw1T    = (_Float16*)alloc(512L * 256 * 2);
  _Float16* mw2T    = (_Float16*)alloc(512L * 512 * 2);
  _Float16* mw3T    = (_Float16*)alloc(128L * 512 * 2);
  // activations
  float*    h0_32   = (float*)alloc((long)NNODE * DD * 4);
  _Float16* h0_16   = (_Float16*)alloc((long)NNODE * DD * 2);
  float*    ping32  = (float*)alloc((long)NNODE * DD * 4);
  _Float16* ping16  = (_Float16*)alloc((long)NNODE * DD * 2);
  float*    pong32  = (float*)alloc((long)NNODE * DD * 4);
  _Float16* pong16  = (_Float16*)alloc((long)NNODE * DD * 2);
  float*    agg32   = (float*)alloc((long)NNODE * DD * 4);
  _Float16* agg16   = (_Float16*)alloc((long)NNODE * DD * 2);
  _Float16* br16[3];
  for (int i = 0; i < 3; ++i) br16[i] = (_Float16*)alloc((long)NNODE * DD * 2);
  _Float16* x016    = (_Float16*)alloc((long)NB * LP * DD * 2);
  float*    conv32  = (float*)alloc((long)NB * L1C * DD * 4);
  _Float16* s116    = (_Float16*)alloc(((long)NB * L1C + 128) * DD * 2);
  _Float16* s216    = (_Float16*)alloc(((long)NB * L2C + 128) * DD * 2);
  _Float16* vpo16   = (_Float16*)alloc(((long)NB * L3C + 64) * DD * 2);
  float*    stats   = (float*)alloc(2L * DD * 4);
  _Float16* q16     = (_Float16*)alloc((long)NB * L3P * HKC * 2);   // [NB,896,768], rows>=885 zero
  _Float16* qT16    = (_Float16*)alloc((long)NB * HKC * L3P * 2);   // [NB,768,896], cols>=885 zero
  float*    v32     = (float*)alloc((long)NNODE * HKC * 4);
  _Float16* vh16    = (_Float16*)alloc(((long)NNODE + 64) * HKC * 2);
  _Float16* att16   = (_Float16*)alloc(((long)NB * NPG + 64) * L3P * 2); // lda/ldc = 896
  float*    T32     = (float*)alloc((long)NB * NPG * HKC * 4);
  float*    fbuf    = (float*)alloc((long)NB * HKC * 4);
  float*    f2buf   = (float*)alloc((long)NB * HDC * 4);
  _Float16* fbn16   = (_Float16*)alloc((long)NB * HDC * 2);
  float*    g32     = (float*)alloc((long)NB * 512 * 4);
  _Float16* gbn16   = (_Float16*)alloc((long)NB * 512 * 2);
  float*    g3_32   = (float*)alloc((long)NB * DD * 4);
  float*    g3bn32  = (float*)alloc((long)NB * DD * 4);
  float*    hsum    = (float*)alloc(HKC * 4);
  float*    hbsum   = (float*)alloc(4);
  float*    scores  = (float*)alloc(3L * NB * 4);

  auto tr = [&](const float* s, _Float16* d, int K, int N, int Kp, int Np) {
    k_transpose_pad<<<cdiv((long)Np * Kp, 256), 256, 0, stream>>>(s, d, K, N, Kp, Np);
  };
  // ---- weight prep: transpose everything to NT Bt layout ----
  k_cvt_pad<<<cdiv((long)NNODE * INFP, 256), 256, 0, stream>>>(node_h, nodeh16, INF_, INFP,
                                                               (long)NNODE * INFP);
  tr(W_init, WinitT, INF_, DD, INFP, DD);
  for (int wl = 0; wl < 9; ++wl) {
    tr(gcn_W + (long)wl * DD * DD, gcnWT + (long)wl * DD * DD, DD, DD, DD, DD);
    tr(gcn_Wres + (long)wl * DD * DD, gcnWrT + (long)wl * DD * DD, DD, DD, DD, DD);
  }
  tr(Wv, WvT, DD, HKC, DD, HKC);
  tr(Wq, WqT, DD, HKC, DD, HKC);
  tr(m_w1, mw1T, 256, 512, 256, 512);
  tr(m_w2, mw2T, 512, 512, 512, 512);
  tr(m_w3, mw3T, 512, 128, 512, 128);
  k_convw<<<cdiv(3L * DD * DD, 256), 256, 0, stream>>>(conv1_w, cwT, 3);
  k_convw<<<cdiv(6L * DD * DD, 256), 256, 0, stream>>>(conv2_w, cwT + 3L * DD * DD, 6);
  k_convw<<<cdiv(9L * DD * DD, 256), 256, 0, stream>>>(conv3_w, cwT + 9L * DD * DD, 9);
  k_hsum<<<cdiv(HKC, 256), 256, 0, stream>>>(h_mat, h_bias, hsum, hbsum);
  k_zero16<<<cdiv((long)NB * L3P * HKC, 256), 256, 0, stream>>>(q16, (long)NB * L3P * HKC);
  k_zero16<<<cdiv(((long)NB * NPG + 64) * L3P, 256), 256, 0, stream>>>(att16,
                                                                       ((long)NB * NPG + 64) * L3P);

  // ---- h0 = node_h @ W_init ----
  gemm(stream, nodeh16, 0, INFP, 0, 0, WinitT, 0, INFP,
       h0_32, h0_16, 0, DD, nullptr, nullptr, nullptr, 0, NNODE, DD, INFP, 1);

  // ---- 3 GCN branches x 3 layers: h = relu(agg@W+b) + relu(h@Wr+br) ----
  const long NH = (long)NNODE * DD;
  for (int b = 0; b < 3; ++b) {
    const float* hc32 = h0_32; const _Float16* hc16 = h0_16;
    float*    o32s[3] = {ping32, pong32, ping32};
    _Float16* o16s[3] = {ping16, pong16, br16[b]};
    for (int l = 0; l < 3; ++l) {
      int wl = b * 3 + l;
      k_zero<<<cdiv(NH, 256), 256, 0, stream>>>(agg32, NH);
      k_scatter<<<cdiv((long)NEDGE * DD, 256), 256, 0, stream>>>(hc32, esrc, edst, agg32,
                                                                 (long)NEDGE * DD);
      k_f32_to_f16<<<cdiv(NH, 256), 256, 0, stream>>>(agg32, agg16, NH);
      gemm(stream, agg16, 0, DD, 0, 0, gcnWT + (long)wl * DD * DD, 0, DD,
           o32s[l], nullptr, 0, DD, gcn_b + wl * DD, nullptr, nullptr, 1, NNODE, DD, DD, 1);
      gemm(stream, hc16, 0, DD, 0, 0, gcnWrT + (long)wl * DD * DD, 0, DD,
           o32s[l], o16s[l], 0, DD, gcn_bres + wl * DD, nullptr, o32s[l], 1, NNODE, DD, DD, 1);
      hc32 = o32s[l]; hc16 = o16s[l];
    }
  }

  // ---- ProteinCNN: embed -> 3x (conv-as-GEMM + bias + relu + BN) ----
  k_embed<<<cdiv((long)NB * LP * DD, 256), 256, 0, stream>>>(v_p, emb, x016);
  gemm(stream, x016, 0, DD, L1C, LP - L1C, cwT, 0, 3 * DD,
       conv32, nullptr, 0, DD, conv_b, nullptr, nullptr, 1, NB * L1C, DD, 3 * DD, 1);
  k_bnchan_stats<<<DD, 256, 0, stream>>>(conv32, stats, NB * L1C);
  k_bnchan_apply<<<cdiv((long)NB * L1C * DD, 256), 256, 0, stream>>>(conv32, stats, cbn_g, cbn_b,
                                                                     s116, (long)NB * L1C * DD);
  gemm(stream, s116, 0, DD, L2C, L1C - L2C, cwT + 3L * DD * DD, 0, 6 * DD,
       conv32, nullptr, 0, DD, conv_b + DD, nullptr, nullptr, 1, NB * L2C, DD, 6 * DD, 1);
  k_bnchan_stats<<<DD, 256, 0, stream>>>(conv32, stats, NB * L2C);
  k_bnchan_apply<<<cdiv((long)NB * L2C * DD, 256), 256, 0, stream>>>(conv32, stats, cbn_g + DD,
                                                                     cbn_b + DD, s216,
                                                                     (long)NB * L2C * DD);
  gemm(stream, s216, 0, DD, L3C, L2C - L3C, cwT + 9L * DD * DD, 0, 9 * DD,
       conv32, nullptr, 0, DD, conv_b + 2 * DD, nullptr, nullptr, 1, NB * L3C, DD, 9 * DD, 1);
  k_bnchan_stats<<<DD, 256, 0, stream>>>(conv32, stats, NB * L3C);
  k_bnchan_apply<<<cdiv((long)NB * L3C * DD, 256), 256, 0, stream>>>(conv32, stats, cbn_g + 2 * DD,
                                                                     cbn_b + 2 * DD, vpo16,
                                                                     (long)NB * L3C * DD);

  // ---- q_ = relu(v_p_o @ Wq + bq), written batched into [NB, 896, 768] (pad rows stay 0) ----
  gemm(stream, vpo16, (long)L3C * DD, DD, 0, 0, WqT, 0, DD,
       nullptr, q16, (long)L3P * HKC, HKC, bq, nullptr, nullptr, 1, L3C, HKC, DD, NB);
  k_qT<<<cdiv((long)NB * HKC * L3P, 256), 256, 0, stream>>>(q16, qT16);

  // ---- BAN + MLP per view (attention linear in heads -> fused batched GEMMs) ----
  for (int vI = 0; vI < 3; ++vI) {
    gemm(stream, br16[vI], 0, DD, 0, 0, WvT, 0, DD,
         v32, nullptr, 0, HKC, bv, nullptr, nullptr, 1, NNODE, HKC, DD, 1);
    k_vh<<<cdiv((long)NNODE * HKC, 256), 256, 0, stream>>>(v32, hsum, vh16, (long)NNODE * HKC);
    // att_sum[b] = (v_*hsum) @ q_[b]^T + sum(h_bias): Bt = q16[b] ([896,768])
    gemm(stream, vh16, (long)NPG * HKC, HKC, 0, 0, q16, (long)L3P * HKC, HKC,
         nullptr, att16, (long)NPG * L3P, L3P, nullptr, hbsum, nullptr, 0, NPG, L3C, HKC, NB);
    // T[b] = att_sum[b] @ q_[b]: Bt = qT16[b] ([768,896]); K padded to 896 (zero cols)
    gemm(stream, att16, (long)NPG * L3P, L3P, 0, 0, qT16, (long)HKC * L3P, L3P,
         T32, nullptr, (long)NPG * HKC, HKC, nullptr, nullptr, nullptr, 0, NPG, HKC, L3P, NB);
    k_fred<<<cdiv((long)NB * HKC, 256), 256, 0, stream>>>(v32, T32, fbuf);
    k_group3<<<cdiv((long)NB * HDC, 256), 256, 0, stream>>>(fbuf, f2buf);
    k_bnfeat<<<1, 256, 0, stream>>>(f2buf, ban_g, ban_b, nullptr, fbn16, HDC);
    // MLP
    gemm(stream, fbn16, 0, HDC, 0, 0, mw1T, 0, HDC,
         g32, nullptr, 0, 512, m_b1, nullptr, nullptr, 1, NB, 512, HDC, 1);
    k_bnfeat<<<2, 256, 0, stream>>>(g32, mg1, mb1, nullptr, gbn16, 512);
    gemm(stream, gbn16, 0, 512, 0, 0, mw2T, 0, 512,
         g32, nullptr, 0, 512, m_b2, nullptr, nullptr, 1, NB, 512, 512, 1);
    k_bnfeat<<<2, 256, 0, stream>>>(g32, mg2, mb2, nullptr, gbn16, 512);
    gemm(stream, gbn16, 0, 512, 0, 0, mw3T, 0, 512,
         g3_32, nullptr, 0, DD, m_b3, nullptr, nullptr, 1, NB, DD, 512, 1);
    k_bnfeat<<<1, 128, 0, stream>>>(g3_32, mg3, mb3, g3bn32, nullptr, DD);
    k_dot_out<<<1, 32, 0, stream>>>(g3bn32, m_w4, m_b4, scores + vI * NB);
  }

  // ---- cosine kernel + CE loss + pack [score_o | s1 | s2 | loss] ----
  k_final<<<1, 32, 0, stream>>>(scores, out);
}
